// SwinTransformerBlock_65146063946593
// MI455X (gfx1250) — matrix operand order repaired and checked by reading, not compiled, not run on hardware
//
#include <hip/hip_runtime.h>

typedef __attribute__((ext_vector_type(16))) _Float16 v16h;
typedef __attribute__((ext_vector_type(8)))  _Float16 v8h;
typedef __attribute__((ext_vector_type(8)))  float    v8f;

#define WMMA_F32_F16(a, b, c) \
  __builtin_amdgcn_wmma_f32_16x16x32_f16(false, (a), false, (b), (short)0, (c), false, false)

constexpr int NTOK  = 49;            // tokens per window
constexpr int DIMC  = 384;
constexpr int HEADS = 12;
constexpr int HD    = 32;            // head dim
constexpr int TBL   = 169;           // (2*7-1)^2
constexpr int NWIN  = 64;            // windows per image
constexpr int BWIN  = 2048;          // B_ = batch * windows
constexpr int MTOT  = BWIN * NTOK;   // 100352 tokens (divisible by 64)
constexpr int BH    = BWIN * HEADS;  // 24576 (window, head) pairs

// ---------------- LDS fragment loaders (contiguous k = 16h + s order, wave32) ----------------
// All LDS tiles are 16-byte aligned; fragment reads are contiguous 16B chunks.

// A fragment 16x32 f16 from row-major f16 LDS [*, lda].
// lanes 0-15: M=lane, halves K 0..15; lanes 16-31: K 16..31.
__device__ __forceinline__ v16h a_frag_lds(const _Float16* A, int lda,
                                           int row0, int k0, int lane) {
  int m  = lane & 15;
  int kb = k0 + ((lane & 16) ? 16 : 0);
  const _Float16* p = A + (row0 + m) * lda + kb;
  v8h lo = *(const v8h*)p;
  v8h hi = *(const v8h*)(p + 8);
  v16h a;
#pragma unroll
  for (int i = 0; i < 8; ++i) { a[i] = lo[i]; a[i + 8] = hi[i]; }
  return a;
}

// B fragment 32x16 f16 where B(k,n) = W[n*32 + k]  (A @ W^T), W tile in LDS [64][32].
// lanes 0-15: N=lane, halves K 0..15; lanes 16-31: K 16..31.
__device__ __forceinline__ v16h bT_frag_lds(const _Float16* W, int n0, int lane) {
  int n  = n0 + (lane & 15);
  int kb = (lane & 16) ? 16 : 0;
  const _Float16* p = W + n * 32 + kb;
  v8h lo = *(const v8h*)p;
  v8h hi = *(const v8h*)(p + 8);
  v16h b;
#pragma unroll
  for (int i = 0; i < 8; ++i) { b[i] = lo[i]; b[i + 8] = hi[i]; }
  return b;
}

// B fragment 32x16 f16, row-major B(k,n) = Bm[k*ldb + n] (K-major stride -> scalar)
__device__ __forceinline__ v16h b_frag_lds_rm(const _Float16* Bm, int ldb,
                                              int k0, int n0, int lane) {
  int n  = n0 + (lane & 15);
  int kb = k0 + ((lane & 16) ? 16 : 0);
  v16h b;
#pragma unroll
  for (int i = 0; i < 16; ++i) b[i] = Bm[(kb + i) * ldb + n];
  return b;
}

// Cooperative stage of a 64x32 fp32 tile -> f16 LDS tile (128 threads).
// thread t: row = t>>1, half-row offset = (t&1)*16.
__device__ __forceinline__ void stage64x32(const float* __restrict__ G, int ldg,
                                           int row0, int k0, _Float16* S, int tid) {
  int r    = tid >> 1;
  int koff = (tid & 1) * 16;
  const float* src = G + (size_t)(row0 + r) * ldg + k0 + koff;
  _Float16* dst = S + r * 32 + koff;
#pragma unroll
  for (int q = 0; q < 4; ++q) {
    float4 f = ((const float4*)src)[q];
    dst[4 * q + 0] = (_Float16)f.x;
    dst[4 * q + 1] = (_Float16)f.y;
    dst[4 * q + 2] = (_Float16)f.z;
    dst[4 * q + 3] = (_Float16)f.w;
  }
}

// ---------------- kernel 1: QKV GEMM (64x64 block tile) + head-major f16 repack ----
__global__ void __launch_bounds__(128) k_qkv(const float* __restrict__ x,
                                             const float* __restrict__ w,
                                             const float* __restrict__ bias,
                                             _Float16* __restrict__ qf,
                                             _Float16* __restrict__ kf,
                                             _Float16* __restrict__ vf) {
  __shared__ __align__(16) _Float16 As[64 * 32];
  __shared__ __align__(16) _Float16 Bs[64 * 32];
  int tid = threadIdx.x, wave = tid >> 5, lane = tid & 31;
  int row0 = blockIdx.x * 64;
  int col0 = blockIdx.y * 64;
  v8f c[4] = {{}, {}, {}, {}};
#pragma unroll 1
  for (int k0 = 0; k0 < DIMC; k0 += 32) {
    stage64x32(x, DIMC, row0, k0, As, tid);
    stage64x32(w, DIMC, col0, k0, Bs, tid);
    if (k0 + 32 < DIMC) {   // prefetch next K-step (global_prefetch_b8)
      __builtin_prefetch(x + (size_t)(row0 + (tid >> 1)) * DIMC + k0 + 32 + (tid & 1) * 16, 0, 1);
      __builtin_prefetch(w + (size_t)(col0 + (tid >> 1)) * DIMC + k0 + 32 + (tid & 1) * 16, 0, 1);
    }
    __syncthreads();
    v16h a = a_frag_lds(As, 32, wave * 16, 0, lane);
#pragma unroll
    for (int j = 0; j < 4; ++j) {
      v16h b = bT_frag_lds(Bs, j * 16, lane);
      c[j] = WMMA_F32_F16(a, b, c[j]);
    }
    __syncthreads();
  }
  // epilogue: bias, q-scale, split to head-major f16 q/k/v
  int rb  = row0 + wave * 16 + ((lane & 16) ? 8 : 0);
  int bw0 = rb / NTOK;
  int tok0 = rb - bw0 * NTOK;
#pragma unroll
  for (int j = 0; j < 4; ++j) {
    int col   = col0 + j * 16 + (lane & 15);
    int which = col / DIMC;              // 0=q 1=k 2=v (16-wide tiles never straddle)
    int ch    = col - which * DIMC;
    int head  = ch >> 5;
    int d     = ch & 31;
    float scl  = (which == 0) ? 0.17677669529663687f : 1.0f;   // hd^-0.5
    float bval = bias[col];
    _Float16* dst = (which == 0) ? qf : ((which == 1) ? kf : vf);
#pragma unroll
    for (int i = 0; i < 8; ++i) {
      int bw = bw0, tok = tok0 + i;
      if (tok >= NTOK) { tok -= NTOK; bw += 1; }   // at most one wrap (8 < 49)
      float val = (c[j][i] + bval) * scl;
      dst[(((size_t)bw * HEADS + head) * NTOK + tok) * HD + d] = (_Float16)val;
    }
  }
}

// ---------------- kernel 2: contextual q/k bias tables ----------------
__global__ void k_biasqk(const _Float16* __restrict__ qf,
                         const _Float16* __restrict__ kf,
                         const float* __restrict__ q_rpe,
                         const float* __restrict__ k_rpe,
                         float* __restrict__ biasqk) {
  int bh = blockIdx.x;
  int h  = bh % HEADS;
  int t  = threadIdx.x;
  if (t >= TBL) return;
  const _Float16* q0 = qf + (size_t)bh * NTOK * HD;   // token 0
  const _Float16* k0 = kf + (size_t)bh * NTOK * HD;
  const float* qr = q_rpe + (size_t)h * HD * TBL;
  const float* kr = k_rpe + (size_t)h * HD * TBL;
  float s = 0.f;
#pragma unroll 4
  for (int d = 0; d < HD; ++d)
    s += (float)q0[d] * qr[d * TBL + t] + (float)k0[d] * kr[d * TBL + t];
  biasqk[(size_t)bh * TBL + t] = s;
}

// ---------------- kernel 3: per-(window,head) attention ----------------
__global__ void __launch_bounds__(128) k_attn(
    const _Float16* __restrict__ qf, const _Float16* __restrict__ kf,
    const _Float16* __restrict__ vf,
    const float* __restrict__ biasqk, const float* __restrict__ mask,
    const int* __restrict__ rpe_idx, const float* __restrict__ v_rpe,
    float* __restrict__ attn_out, float* __restrict__ out_pre) {
  __shared__ __align__(16) _Float16 qs[64 * HD];
  __shared__ __align__(16) _Float16 ks[64 * HD];
  __shared__ __align__(16) _Float16 vs[64 * HD];
  __shared__ __align__(16) float    Sf[64 * 64];
  __shared__ __align__(16) _Float16 Sh[64 * 64];
  __shared__ __align__(16) float    o1[64 * HD];
  __shared__ unsigned char idx8[NTOK * NTOK];

  int tid  = threadIdx.x;
  int wave = tid >> 5;
  int lane = tid & 31;
  int bh   = blockIdx.x;
  int b    = bh / HEADS;
  int h    = bh - b * HEADS;
  int win  = b % NWIN;

  const v8h* qg = (const v8h*)(qf + (size_t)bh * NTOK * HD);  // 3136B aligned
  const v8h* kg = (const v8h*)(kf + (size_t)bh * NTOK * HD);
  const v8h* vg = (const v8h*)(vf + (size_t)bh * NTOK * HD);
  for (int vec = tid; vec < 256; vec += 128) {               // 2048 halves; 196 vecs valid
    v8h z = {};
    bool ok = vec < (NTOK * HD) / 8;
    ((v8h*)qs)[vec] = ok ? qg[vec] : z;
    ((v8h*)ks)[vec] = ok ? kg[vec] : z;
    ((v8h*)vs)[vec] = ok ? vg[vec] : z;
  }
  for (int i = tid; i < NTOK * NTOK; i += 128) idx8[i] = (unsigned char)rpe_idx[i];
  __syncthreads();

  // ---- S = q @ k^T : each wave owns 16 rows x 64 cols (4 WMMAs, K=hd=32) ----
  int m0 = wave * 16;
  {
    v16h aq = a_frag_lds(qs, HD, m0, 0, lane);
    const float* bb = biasqk + (size_t)bh * TBL;
    const float* mk = mask + (size_t)win * NTOK * NTOK;
#pragma unroll 1
    for (int jt = 0; jt < 4; ++jt) {
      v16h bk = bT_frag_lds(ks, jt * 16, lane);   // B(k=d,n=j) = ks[j][d]
      v8f cc = {};
      cc = WMMA_F32_F16(aq, bk, cc);
      int j  = jt * 16 + (lane & 15);
      int rb = m0 + ((lane & 16) ? 8 : 0);
#pragma unroll
      for (int i2 = 0; i2 < 8; ++i2) {
        int i = rb + i2;
        float s = -1e30f;
        if (i < NTOK && j < NTOK)
          s = cc[i2] + bb[idx8[i * NTOK + j]] + mk[i * NTOK + j];
        Sf[i * 64 + j] = s;
      }
    }
  }
  __syncthreads();

  // ---- softmax (one thread per row), write attn to d_out ----
  if (tid < 64) {
    int i = tid;
    float* row = Sf + i * 64;
    if (i < NTOK) {
      float mx = -1e30f;
      for (int j = 0; j < NTOK; ++j) mx = fmaxf(mx, row[j]);
      float sum = 0.f;
      for (int j = 0; j < NTOK; ++j) { float e = __expf(row[j] - mx); row[j] = e; sum += e; }
      float inv = 1.f / sum;
      float* ao = attn_out + ((size_t)bh * NTOK + i) * NTOK;
      for (int j = 0; j < 64; ++j) {
        float a = (j < NTOK) ? row[j] * inv : 0.f;
        row[j] = a;
        Sh[i * 64 + j] = (_Float16)a;
        if (j < NTOK) ao[j] = a;
      }
    } else {
      for (int j = 0; j < 64; ++j) { row[j] = 0.f; Sh[i * 64 + j] = (_Float16)0.f; }
    }
  }
  __syncthreads();

  // ---- o1 = attn @ v : M=64 (4 waves), N=32 (2 tiles), K=64 (2 steps) ----
#pragma unroll 1
  for (int nt = 0; nt < 2; ++nt) {
    v8f cc = {};
#pragma unroll 1
    for (int k0 = 0; k0 < 64; k0 += 32) {
      v16h a  = a_frag_lds(Sh, 64, m0, k0, lane);
      v16h bv = b_frag_lds_rm(vs, HD, k0, nt * 16, lane);   // B(k=j,n=d) = vs[j][d]
      cc = WMMA_F32_F16(a, bv, cc);
    }
    int dcol = nt * 16 + (lane & 15);
    int rb   = m0 + ((lane & 16) ? 8 : 0);
#pragma unroll
    for (int i2 = 0; i2 < 8; ++i2) o1[(rb + i2) * HD + dcol] = cc[i2];
  }
  __syncthreads();

  // ---- gathered v-rpe bias (i-dependent weights => VALU) + combine ----
  const float* vr = v_rpe + (size_t)h * TBL * HD;
  for (int e = tid; e < NTOK * HD; e += 128) {
    int i = e >> 5, d = e & 31;
    const float* srow = Sf + i * 64;
    const unsigned char* irow = idx8 + i * NTOK;
    float acc = 0.f;
    for (int j = 0; j < NTOK; ++j) acc += srow[j] * vr[(int)irow[j] * HD + d];
    out_pre[((size_t)b * NTOK + i) * DIMC + h * HD + d] = o1[i * HD + d] + acc;
  }
}

// ---------------- kernel 4: output projection GEMM (64x64 block tile) ----------------
__global__ void __launch_bounds__(128) k_proj(const float* __restrict__ A,
                                              const float* __restrict__ w,
                                              const float* __restrict__ bias,
                                              float* __restrict__ out) {
  __shared__ __align__(16) _Float16 As[64 * 32];
  __shared__ __align__(16) _Float16 Bs[64 * 32];
  int tid = threadIdx.x, wave = tid >> 5, lane = tid & 31;
  int row0 = blockIdx.x * 64;
  int col0 = blockIdx.y * 64;
  v8f c[4] = {{}, {}, {}, {}};
#pragma unroll 1
  for (int k0 = 0; k0 < DIMC; k0 += 32) {
    stage64x32(A, DIMC, row0, k0, As, tid);
    stage64x32(w, DIMC, col0, k0, Bs, tid);
    if (k0 + 32 < DIMC) {
      __builtin_prefetch(A + (size_t)(row0 + (tid >> 1)) * DIMC + k0 + 32 + (tid & 1) * 16, 0, 1);
      __builtin_prefetch(w + (size_t)(col0 + (tid >> 1)) * DIMC + k0 + 32 + (tid & 1) * 16, 0, 1);
    }
    __syncthreads();
    v16h a = a_frag_lds(As, 32, wave * 16, 0, lane);
#pragma unroll
    for (int j = 0; j < 4; ++j) {
      v16h b = bT_frag_lds(Bs, j * 16, lane);
      c[j] = WMMA_F32_F16(a, b, c[j]);
    }
    __syncthreads();
  }
  int rb = row0 + wave * 16 + ((lane & 16) ? 8 : 0);
#pragma unroll
  for (int j = 0; j < 4; ++j) {
    int col = col0 + j * 16 + (lane & 15);
    float bval = bias[col];
#pragma unroll
    for (int i = 0; i < 8; ++i)
      out[(size_t)(rb + i) * DIMC + col] = c[j][i] + bval;
  }
}

// ---------------- launcher ----------------
extern "C" void kernel_launch(void* const* d_in, const int* in_sizes, int n_in,
                              void* d_out, int out_size, void* d_ws, size_t ws_size,
                              hipStream_t stream) {
  (void)in_sizes; (void)n_in; (void)out_size; (void)ws_size;
  const float* x      = (const float*)d_in[0];
  const float* mask   = (const float*)d_in[1];
  const int*   rpe    = (const int*)d_in[2];
  const float* qkv_w  = (const float*)d_in[3];
  const float* qkv_b  = (const float*)d_in[4];
  const float* proj_w = (const float*)d_in[5];
  const float* proj_b = (const float*)d_in[6];
  const float* q_rpe  = (const float*)d_in[7];
  const float* k_rpe  = (const float*)d_in[8];
  const float* v_rpe  = (const float*)d_in[9];

  float* out  = (float*)d_out;                     // (B_, N, DIM)
  float* attn = out + (size_t)MTOT * DIMC;         // (B_, H, N, N) appended

  char* ws = (char*)d_ws;
  size_t qkvb = (size_t)BH * NTOK * HD * sizeof(_Float16);
  _Float16* qf = (_Float16*)ws;  ws += qkvb;
  _Float16* kf = (_Float16*)ws;  ws += qkvb;
  _Float16* vf = (_Float16*)ws;  ws += qkvb;
  float* biasqk = (float*)ws;    ws += (size_t)BH * TBL * sizeof(float);
  float* outpre = (float*)ws;    // (B_, N, DIM) fp32

  dim3 g1(MTOT / 64, (3 * DIMC) / 64);
  k_qkv<<<g1, 128, 0, stream>>>(x, qkv_w, qkv_b, qf, kf, vf);
  k_biasqk<<<BH, 192, 0, stream>>>(qf, kf, q_rpe, k_rpe, biasqk);
  k_attn<<<BH, 128, 0, stream>>>(qf, kf, vf, biasqk, mask, rpe, v_rpe, attn, outpre);
  dim3 g3(MTOT / 64, DIMC / 64);
  k_proj<<<g3, 128, 0, stream>>>(outpre, proj_w, proj_b, out);
}
